// MultiheadIsotropicAFA_63299228008886
// MI455X (gfx1250) — compile-verified, matrix operand order checked by reading, not run on hardware
//
#include <hip/hip_runtime.h>
#include <hip/hip_bf16.h>

// MI455X / gfx1250: wave32, WMMA 16x16x32 bf16 (f32 accum).
// All WMMA operands stored bf16 in global (half the HBM traffic); GEMM stages
// tiles into double-buffered LDS; A and N x K B as a pure byte copy — via the CDNA5 async
// global->LDS path (GLOBAL_LOAD_ASYNC_TO_LDS_B128, ASYNCcnt) when the
// toolchain exposes the builtins, otherwise via batched b128 register copies.
// Fragments load as 2x ds_load_b128 per 16x32 fragment (N-major B in LDS).
//
// Pipeline (B=2, L=1024, DE=1024, H=8, DK=DV=64):
//   0) fp32 -> bf16 conversion of Z_{q,k,v} and W_{q,k,v,o}
//   1) Q/K/V projections : (2048x1024)@(1024x1024)+bias      [WMMA GEMM]
//   2) rotate Q,K,V -> bf16; row norms qn,kn                  [VALU]
//   3) dot = Qr @ Kr^T   : 16 x (1024x128)@(128x1024)         [WMMA GEMM, async A+B]
//   4) log-score softmax + decay -> A_hat (bf16)              [VALU]
//   5) est = A_hat @ Vr  : 16 x (1024x1024)@(1024x128)        [WMMA GEMM]
//   6) un-rotate + reorder -> Xo (bf16)                        [VALU]
//   7) out = Xo @ Wo + b : (2048x1024)@(1024x1024)+bias       [WMMA GEMM]

typedef __attribute__((ext_vector_type(16))) __bf16 v16bf;
typedef __attribute__((ext_vector_type(8)))  __bf16 v8bf;
typedef __attribute__((ext_vector_type(2)))  __bf16 v2bf;
typedef __attribute__((ext_vector_type(8)))  float  v8f;
typedef __attribute__((ext_vector_type(4)))  int    v4i;

#define TILE_BM 128
#define TILE_BN 128
#define TILE_BK 32
#define LDS_PAD 8   // row stride = 40 bf16 = 80 B, multiple of 16 B

// CDNA5 async global->LDS copy (GLOBAL_LOAD_ASYNC_TO_LDS_B128, ASYNCcnt),
// guarded so the kernel still compiles on toolchains without the builtins.
// Probe-confirmed signature: (v4i AS1* gsrc, v4i AS3* ldst, imm, imm).
#if defined(__has_builtin)
#if __has_builtin(__builtin_amdgcn_global_load_async_to_lds_b128) && \
    __has_builtin(__builtin_amdgcn_s_wait_asynccnt)
#define USE_ASYNC_COPY 1
#endif
#endif
#ifndef USE_ASYNC_COPY
#define USE_ASYNC_COPY 0
#endif

#define AS_GLOBAL __attribute__((address_space(1)))
#define AS_SHARED __attribute__((address_space(3)))

// 16-bit 16x32 fragment (CDNA5 ISA 7.12.2): lane half 0 holds K={0..7,16..23},
// half 1 holds K={8..15,24..31}; element e<8 -> K=half*8+e, e>=8 -> K=16+half*8+(e-8).
// With a K-contiguous LDS row this is two aligned 16-byte loads.
__device__ __forceinline__ v16bf load_frag(const __bf16* rowp, int half) {
    const v8bf lo = *reinterpret_cast<const v8bf*>(rowp + half * 8);
    const v8bf hi = *reinterpret_cast<const v8bf*>(rowp + 16 + half * 8);
    return __builtin_shufflevector(lo, hi, 0, 1, 2, 3, 4, 5, 6, 7,
                                           8, 9, 10, 11, 12, 13, 14, 15);
}

template <bool TRANS_B>
__global__ __launch_bounds__(256)
void wmma_gemm_kernel(const __bf16* __restrict__ A, const __bf16* __restrict__ Bm,
                      const float* __restrict__ bias, float* __restrict__ C,
                      int M, int N, int K,
                      long long sA, long long sB, long long sC)
{
    (void)M;
    // Double-buffered tiles; Bs is N-major ([n][k]) for fragment-contiguous reads.
    __shared__ __bf16 As[2][TILE_BM][TILE_BK + LDS_PAD];
    __shared__ __bf16 Bs[2][TILE_BN][TILE_BK + LDS_PAD];

    const int tid  = threadIdx.x;
    const int lane = tid & 31;          // wave32
    const int wave = tid >> 5;          // 8 waves
    const int wm   = wave & 3;          // 4 waves along M -> 32 rows each
    const int wn   = wave >> 2;         // 2 waves along N -> 64 cols each
    const int half = lane >> 4;
    const int r16  = lane & 15;

    const __bf16* Ab = A  + (size_t)blockIdx.z * sA;
    const __bf16* Bb = Bm + (size_t)blockIdx.z * sB;
    float*        Cb = C  + (size_t)blockIdx.z * sC;

    const int row0 = blockIdx.y * TILE_BM;
    const int col0 = blockIdx.x * TILE_BN;

    // A tile: 128x32 bf16 = 512 x 16B chunks, 2 per thread (pure copy).
    auto stageA = [&](int buf, int k0) {
#if USE_ASYNC_COPY
        #pragma unroll
        for (int i = 0; i < 2; ++i) {
            const int idx = tid + i * 256;
            const int r = idx >> 2, c = (idx & 3) << 3;
            __builtin_amdgcn_global_load_async_to_lds_b128(
                (AS_GLOBAL v4i*)(Ab + (size_t)(row0 + r) * K + k0 + c),
                (AS_SHARED v4i*)&As[buf][r][c],
                0, 0);
        }
#else
        v8bf t[2];
        #pragma unroll
        for (int i = 0; i < 2; ++i) {
            const int idx = tid + i * 256;
            t[i] = *reinterpret_cast<const v8bf*>(
                Ab + (size_t)(row0 + (idx >> 2)) * K + k0 + ((idx & 3) << 3));
        }
        #pragma unroll
        for (int i = 0; i < 2; ++i) {
            const int idx = tid + i * 256;
            *reinterpret_cast<v8bf*>(&As[buf][idx >> 2][(idx & 3) << 3]) = t[i];
        }
#endif
    };

    auto stageB = [&](int buf, int k0) {
        if (TRANS_B) {
            // B is (N x K) row-major: pure copy into Bs[n][k].
#if USE_ASYNC_COPY
            #pragma unroll
            for (int i = 0; i < 2; ++i) {
                const int idx = tid + i * 256;
                const int n = idx >> 2, c = (idx & 3) << 3;
                __builtin_amdgcn_global_load_async_to_lds_b128(
                    (AS_GLOBAL v4i*)(Bb + (size_t)(col0 + n) * K + k0 + c),
                    (AS_SHARED v4i*)&Bs[buf][n][c],
                    0, 0);
            }
#else
            v8bf t[2];
            #pragma unroll
            for (int i = 0; i < 2; ++i) {
                const int idx = tid + i * 256;
                t[i] = *reinterpret_cast<const v8bf*>(
                    Bb + (size_t)(col0 + (idx >> 2)) * K + k0 + ((idx & 3) << 3));
            }
            #pragma unroll
            for (int i = 0; i < 2; ++i) {
                const int idx = tid + i * 256;
                *reinterpret_cast<v8bf*>(&Bs[buf][idx >> 2][(idx & 3) << 3]) = t[i];
            }
#endif
        } else {
            // B is (K x N) row-major: read 8 n-contiguous bf16, transpose into Bs[n][k].
            v8bf t[2];
            #pragma unroll
            for (int i = 0; i < 2; ++i) {
                const int idx = tid + i * 256;
                const int r = idx >> 4;            // k: 0..31
                const int c = (idx & 15) << 3;     // n: 0..120
                t[i] = *reinterpret_cast<const v8bf*>(
                    Bb + (size_t)(k0 + r) * N + col0 + c);
            }
            #pragma unroll
            for (int i = 0; i < 2; ++i) {
                const int idx = tid + i * 256;
                const int r = idx >> 4;
                const int c = (idx & 15) << 3;
                #pragma unroll
                for (int e = 0; e < 8; ++e) Bs[buf][c + e][r] = t[i][e];
            }
        }
    };

    v8f acc[2][4] = {};

    const int nk = K / TILE_BK;
    stageA(0, 0);
    stageB(0, 0);
#if USE_ASYNC_COPY
    __builtin_amdgcn_s_wait_asynccnt(0);
#endif
    __syncthreads();

    for (int kt = 0; kt < nk; ++kt) {
        const int buf = kt & 1;
        const int nxt = kt + 1;

        if (nxt + 1 < nk) {  // far prefetch (global_prefetch_b8)
            __builtin_prefetch(Ab + (size_t)(row0 + (tid >> 1)) * K
                                  + (nxt + 1) * TILE_BK, 0, 0);
        }
        if (nxt < nk) {      // stage next tile into the other buffer
            stageA(nxt & 1, nxt * TILE_BK);
            stageB(nxt & 1, nxt * TILE_BK);
        }

        // Fragment loads: 2 x ds_load_b128 per fragment.
        v16bf afrag[2], bfrag[4];
        #pragma unroll
        for (int mt = 0; mt < 2; ++mt)
            afrag[mt] = load_frag(&As[buf][wm * 32 + mt * 16 + r16][0], half);
        #pragma unroll
        for (int nt = 0; nt < 4; ++nt)
            bfrag[nt] = load_frag(&Bs[buf][wn * 64 + nt * 16 + r16][0], half);

        // 2x4 WMMA tiles per wave: D = A x B + C (f32 accumulate).
        #pragma unroll
        for (int mt = 0; mt < 2; ++mt)
            #pragma unroll
            for (int nt = 0; nt < 4; ++nt)
                acc[mt][nt] = __builtin_amdgcn_wmma_f32_16x16x32_bf16(
                    false, afrag[mt], false, bfrag[nt],
                    (short)0, acc[mt][nt], false, false);

#if USE_ASYNC_COPY
        __builtin_amdgcn_s_wait_asynccnt(0);
#endif
        __syncthreads();
    }

    // Epilogue: C/D layout — VGPR v: lanes 0-15 -> M=v, lanes 16-31 -> M=v+8; N=lane%16.
    #pragma unroll
    for (int mt = 0; mt < 2; ++mt) {
        const int cm = row0 + wm * 32 + mt * 16 + half * 8;
        #pragma unroll
        for (int nt = 0; nt < 4; ++nt) {
            const int cn = col0 + wn * 64 + nt * 16 + r16;
            const float bv = bias ? bias[cn] : 0.0f;
            #pragma unroll
            for (int v = 0; v < 8; ++v) {
                Cb[(size_t)(cm + v) * N + cn] = acc[mt][nt][v] + bv;
            }
        }
    }
}

// Bulk fp32 -> bf16 conversion, 8 elements per thread (b128 in, b128 out).
__global__ __launch_bounds__(256)
void f32_to_bf16_kernel(const float* __restrict__ src, __bf16* __restrict__ dst)
{
    const size_t i = (size_t)blockIdx.x * 256 + threadIdx.x;
    const float4 a = reinterpret_cast<const float4*>(src)[2 * i];
    const float4 b = reinterpret_cast<const float4*>(src)[2 * i + 1];
    v8bf o;
    o[0] = (__bf16)a.x; o[1] = (__bf16)a.y; o[2] = (__bf16)a.z; o[3] = (__bf16)a.w;
    o[4] = (__bf16)b.x; o[5] = (__bf16)b.y; o[6] = (__bf16)b.z; o[7] = (__bf16)b.w;
    reinterpret_cast<v8bf*>(dst)[i] = o;
}

// Rotate Q,K,V by theta = -t*omega (complex pairs), emit bf16 (B,H,L,128)
// layouts with inner index 2*d + c, plus per-(b,h,l) squared norms of Q and K.
__global__ __launch_bounds__(64)
void rotate_qkv_kernel(const float* __restrict__ Qp, const float* __restrict__ Kp,
                       const float* __restrict__ Vp, const float* __restrict__ lam,
                       __bf16* __restrict__ Qr, __bf16* __restrict__ Kr,
                       __bf16* __restrict__ Vr,
                       float* __restrict__ qn, float* __restrict__ kn)
{
    const int d = threadIdx.x;      // 0..63
    const int l = blockIdx.x;       // 0..1023
    const int h = blockIdx.y;       // 0..7
    const int b = blockIdx.z;       // 0..1

    // omega = interleave(lambda_imag, -lambda_imag)
    const float om  = lam[h * 32 + (d >> 1)] * ((d & 1) ? -1.0f : 1.0f);
    const float th  = -(float)l * om;
    const float cth = cosf(th);
    const float sth = sinf(th);

    // Input layout: (B, L, [c=2, H, 64]) -> c stride 512
    const size_t zin = ((size_t)b * 1024 + l) * 1024;
    const int    co  = h * 64 + d;
    const float qre = Qp[zin + co], qim = Qp[zin + 512 + co];
    const float kre = Kp[zin + co], kim = Kp[zin + 512 + co];
    const float vre = Vp[zin + co], vim = Vp[zin + 512 + co];

    const size_t zo = (((size_t)(b * 8 + h)) * 1024 + l) * 128 + 2 * d;
    v2bf q, k, v;
    q[0] = (__bf16)(qre * cth - qim * sth); q[1] = (__bf16)(qre * sth + qim * cth);
    k[0] = (__bf16)(kre * cth - kim * sth); k[1] = (__bf16)(kre * sth + kim * cth);
    v[0] = (__bf16)(vre * cth - vim * sth); v[1] = (__bf16)(vre * sth + vim * cth);
    *reinterpret_cast<v2bf*>(&Qr[zo]) = q;
    *reinterpret_cast<v2bf*>(&Kr[zo]) = k;
    *reinterpret_cast<v2bf*>(&Vr[zo]) = v;

    __shared__ float rq[64], rk[64];
    rq[d] = qre * qre + qim * qim;
    rk[d] = kre * kre + kim * kim;
    __syncthreads();
    #pragma unroll
    for (int s2 = 32; s2 > 0; s2 >>= 1) {
        if (d < s2) { rq[d] += rq[d + s2]; rk[d] += rk[d + s2]; }
        __syncthreads();
    }
    if (d == 0) {
        const size_t o = ((size_t)(b * 8 + h)) * 1024 + l;
        qn[o] = rq[0];
        kn[o] = rk[0];
    }
}

// Per-row (b,h,i): build scores from dot/qn/kn/decay, masked causal softmax,
// multiply by decay, write A_hat as bf16.
__global__ __launch_bounds__(256)
void softmax_kernel(const float* __restrict__ dotA, __bf16* __restrict__ Ahat,
                    const float* __restrict__ qn, const float* __restrict__ kn,
                    const float* __restrict__ mu, const float* __restrict__ sigma,
                    const float* __restrict__ eta, const float* __restrict__ gam,
                    const float* __restrict__ tau, const float* __restrict__ nus)
{
    const int i = blockIdx.x, h = blockIdx.y, b = blockIdx.z;
    const int tid = threadIdx.x;
    const int z = b * 8 + h;

    const float* row  = dotA + (((size_t)z) * 1024 + i) * 1024;
    __bf16*      arow = Ahat + (((size_t)z) * 1024 + i) * 1024;
    const float  qni = qn[(size_t)z * 1024 + i];
    const float* knr = kn + (size_t)z * 1024;

    const float muh   = mu[h];
    const float alpha = muh * muh;
    const float sig2  = sigma[h] * sigma[h];
    const float eta2  = eta[h] * eta[h];
    const float gam2  = gam[h] * gam[h];
    const float tau2  = tau[h] * tau[h];
    const float nsq   = nus[h];
    const float nu    = nsq * nsq;

    float sc[4];
    float lmax = -INFINITY;
    #pragma unroll
    for (int it = 0; it < 4; ++it) {
        const int j = tid + it * 256;
        float s = -INFINITY;
        if (j <= i) {
            const float dt  = (float)(i - j);
            const float dec = expf(-dt * alpha);
            const float d2  = dec * dec;
            const float Vij = sig2 * (1.0f - d2) / (2.0f * alpha + 1e-6f)
                            + eta2 * d2 + gam2;
            const float R2  = fmaxf(qni + d2 * knr[j] - 2.0f * dec * row[j], 0.0f);
            const float Sv  = 1.0f / (nu * 64.0f * Vij);
            s = tau2 * (-logf(Vij) - log1pf(R2 * Sv));
        }
        sc[it] = s;
        lmax = fmaxf(lmax, s);
    }

    __shared__ float red[256];
    red[tid] = lmax;
    __syncthreads();
    for (int t = 128; t > 0; t >>= 1) {
        if (tid < t) red[tid] = fmaxf(red[tid], red[tid + t]);
        __syncthreads();
    }
    const float mx = red[0];
    __syncthreads();

    float pv[4];
    float lsum = 0.0f;
    #pragma unroll
    for (int it = 0; it < 4; ++it) {
        const int j = tid + it * 256;
        const float p = (j <= i) ? expf(sc[it] - mx) : 0.0f;
        pv[it] = p;
        lsum += p;
    }
    red[tid] = lsum;
    __syncthreads();
    for (int t = 128; t > 0; t >>= 1) {
        if (tid < t) red[tid] += red[tid + t];
        __syncthreads();
    }
    const float inv = 1.0f / red[0];

    #pragma unroll
    for (int it = 0; it < 4; ++it) {
        const int j = tid + it * 256;
        float out = 0.0f;
        if (j <= i) {
            const float dec = expf(-(float)(i - j) * alpha);
            out = pv[it] * inv * dec;
        }
        arow[j] = (__bf16)out;
    }
}

// Inverse rotation (theta = +t*omega) and reorder (B,H,L,[d,c]) -> (B,L,[c,H,d]) bf16.
__global__ __launch_bounds__(64)
void unrotate_kernel(const float* __restrict__ est, const float* __restrict__ lam,
                     __bf16* __restrict__ Xo)
{
    const int d = threadIdx.x, l = blockIdx.x, h = blockIdx.y, b = blockIdx.z;
    const float om  = lam[h * 32 + (d >> 1)] * ((d & 1) ? -1.0f : 1.0f);
    const float th  = (float)l * om;
    const float cth = cosf(th);
    const float sth = sinf(th);

    const size_t zi = (((size_t)(b * 8 + h)) * 1024 + l) * 128 + 2 * d;
    const float er = est[zi], ei = est[zi + 1];

    const size_t zo = ((size_t)b * 1024 + l) * 1024;
    const int    co = h * 64 + d;
    Xo[zo + co]       = (__bf16)(er * cth - ei * sth);
    Xo[zo + 512 + co] = (__bf16)(er * sth + ei * cth);
}

extern "C" void kernel_launch(void* const* d_in, const int* in_sizes, int n_in,
                              void* d_out, int out_size, void* d_ws, size_t ws_size,
                              hipStream_t stream)
{
    (void)in_sizes; (void)n_in; (void)out_size; (void)ws_size;

    const float* Zq   = (const float*)d_in[0];
    const float* Zk   = (const float*)d_in[1];
    const float* Zv   = (const float*)d_in[2];
    const float* Wq_w = (const float*)d_in[3];
    const float* Wq_b = (const float*)d_in[4];
    const float* Wk_w = (const float*)d_in[5];
    const float* Wk_b = (const float*)d_in[6];
    const float* Wv_w = (const float*)d_in[7];
    const float* Wv_b = (const float*)d_in[8];
    const float* Wo_w = (const float*)d_in[9];
    const float* Wo_b = (const float*)d_in[10];
    const float* lam  = (const float*)d_in[11];
    const float* mu   = (const float*)d_in[12];
    const float* sig  = (const float*)d_in[13];
    const float* eta  = (const float*)d_in[14];
    const float* gam  = (const float*)d_in[15];
    const float* tau  = (const float*)d_in[16];
    const float* nus  = (const float*)d_in[17];

    // Workspace carving (~152 MB).
    char* w = (char*)d_ws;
    auto allocf = [&](size_t n) { float*  p = (float*)w;  w += n * sizeof(float);  return p; };
    auto allocb = [&](size_t n) { __bf16* p = (__bf16*)w; w += n * sizeof(__bf16); return p; };

    const size_t NPROJ = (size_t)2048 * 1024;          // 2M elements
    const size_t NW    = (size_t)1024 * 1024;          // 1M elements
    const size_t NROT  = (size_t)16 * 1024 * 128;      // 2M elements
    const size_t NROW  = (size_t)16 * 1024;            // 16K elements
    const size_t NATT  = (size_t)16 * 1024 * 1024;     // 16M elements

    float*  Qp   = allocf(NPROJ);      // fp32 projection outputs
    float*  Kp   = allocf(NPROJ);      // reused as est after rotation
    float*  Vp   = allocf(NPROJ);
    float*  dotA = allocf(NATT);
    float*  qn   = allocf(NROW);
    float*  kn   = allocf(NROW);
    __bf16* Zqb  = allocb(NPROJ);      // reused as Xo after stage 6
    __bf16* Zkb  = allocb(NPROJ);
    __bf16* Zvb  = allocb(NPROJ);
    __bf16* Wqb  = allocb(NW);
    __bf16* Wkb  = allocb(NW);
    __bf16* Wvb  = allocb(NW);
    __bf16* Wob  = allocb(NW);
    __bf16* Qr   = allocb(NROT);
    __bf16* Kr   = allocb(NROT);
    __bf16* Vr   = allocb(NROT);
    __bf16* Ahat = allocb(NATT);
    float*  est  = Kp;
    __bf16* Xo   = Zqb;

    const dim3 blk(256);

    // 0) operand conversion to bf16 (one pass each; 8 elem/thread)
    f32_to_bf16_kernel<<<dim3(NPROJ / (256 * 8)), blk, 0, stream>>>(Zq, Zqb);
    f32_to_bf16_kernel<<<dim3(NPROJ / (256 * 8)), blk, 0, stream>>>(Zk, Zkb);
    f32_to_bf16_kernel<<<dim3(NPROJ / (256 * 8)), blk, 0, stream>>>(Zv, Zvb);
    f32_to_bf16_kernel<<<dim3(NW / (256 * 8)), blk, 0, stream>>>(Wq_w, Wqb);
    f32_to_bf16_kernel<<<dim3(NW / (256 * 8)), blk, 0, stream>>>(Wk_w, Wkb);
    f32_to_bf16_kernel<<<dim3(NW / (256 * 8)), blk, 0, stream>>>(Wv_w, Wvb);
    f32_to_bf16_kernel<<<dim3(NW / (256 * 8)), blk, 0, stream>>>(Wo_w, Wob);

    // 1) Q/K/V projections: M=2048, N=1024, K=1024
    wmma_gemm_kernel<false><<<dim3(8, 16, 1), blk, 0, stream>>>(
        Zqb, Wqb, Wq_b, Qp, 2048, 1024, 1024, 0, 0, 0);
    wmma_gemm_kernel<false><<<dim3(8, 16, 1), blk, 0, stream>>>(
        Zkb, Wkb, Wk_b, Kp, 2048, 1024, 1024, 0, 0, 0);
    wmma_gemm_kernel<false><<<dim3(8, 16, 1), blk, 0, stream>>>(
        Zvb, Wvb, Wv_b, Vp, 2048, 1024, 1024, 0, 0, 0);

    // 2) rotation + norms (bf16 out)
    rotate_qkv_kernel<<<dim3(1024, 8, 2), dim3(64), 0, stream>>>(
        Qp, Kp, Vp, lam, Qr, Kr, Vr, qn, kn);

    // 3) dot = Qr @ Kr^T : 16 batches, M=N=1024, K=128 (Kr stored N-major)
    wmma_gemm_kernel<true><<<dim3(8, 8, 16), blk, 0, stream>>>(
        Qr, Kr, nullptr, dotA, 1024, 1024, 128,
        (long long)1024 * 128, (long long)1024 * 128, (long long)1024 * 1024);

    // 4) scores -> softmax -> * decay -> bf16 A_hat
    softmax_kernel<<<dim3(1024, 8, 2), blk, 0, stream>>>(
        dotA, Ahat, qn, kn, mu, sig, eta, gam, tau, nus);

    // 5) est = A_hat @ Vr : 16 batches, M=1024, N=128, K=1024
    wmma_gemm_kernel<false><<<dim3(1, 8, 16), blk, 0, stream>>>(
        Ahat, Vr, nullptr, est, 1024, 128, 1024,
        (long long)1024 * 1024, (long long)1024 * 128, (long long)1024 * 128);

    // 6) inverse rotation + reorder (bf16 out)
    unrotate_kernel<<<dim3(1024, 8, 2), dim3(64), 0, stream>>>(est, lam, Xo);

    // 7) output projection with bias -> d_out (fp32)
    wmma_gemm_kernel<false><<<dim3(8, 16, 1), blk, 0, stream>>>(
        Xo, Wob, Wo_b, (float*)d_out, 2048, 1024, 1024, 0, 0, 0);
}
